// Decoder_49022756717163
// MI455X (gfx1250) — matrix operand order repaired and checked by reading, not compiled, not run on hardware
//
#include <hip/hip_runtime.h>
#include <hip/hip_bf16.h>
#include <math.h>

// ---------------- problem constants (match reference) ----------------
constexpr int DEPTH_ = 4;
constexpr int EMB    = 512;
constexpr int NHEADS = 8;
constexpr int FFDIM  = 2048;
constexpr int NB     = 8;      // batch
constexpr int SEQ    = 1024;   // sequence length
constexpr int HD     = EMB / NHEADS;   // 64
constexpr float LNEPS = 1e-5f;
constexpr float POS_SCALE = 0.1f;

typedef __attribute__((ext_vector_type(16))) __bf16 v16bf;
typedef __attribute__((ext_vector_type(8)))  float  v8f;

#if defined(__has_builtin)
#if __has_builtin(__builtin_amdgcn_tensor_load_to_lds) && __has_builtin(__builtin_amdgcn_s_wait_tensorcnt)
#define HAVE_TDM 1
#endif
#endif
#ifndef HAVE_TDM
#define HAVE_TDM 0
#endif

union FragU  { v16bf v; uint4 u[2]; };
union Half8  { uint4 u; unsigned short s[8]; };

// fp32 -> bf16 round-to-nearest-even as raw u16 (no __bf16 arithmetic)
__device__ __forceinline__ unsigned short f2bf(float f) {
    unsigned int u = __float_as_uint(f);
    unsigned int r = (u + 0x7FFFu + ((u >> 16) & 1u)) >> 16;
    return (unsigned short)r;
}

// ---------------- block reductions (exactly 256 threads = 8 wave32) --
__device__ __forceinline__ float blockReduceSum(float v) {
    __shared__ float sm[8];
    #pragma unroll
    for (int o = 16; o > 0; o >>= 1) v += __shfl_xor(v, o, 32);
    int w = threadIdx.x >> 5;
    __syncthreads();
    if ((threadIdx.x & 31) == 0) sm[w] = v;
    __syncthreads();
    float r = sm[0];
    #pragma unroll
    for (int i = 1; i < 8; i++) r += sm[i];
    return r;
}
__device__ __forceinline__ float blockReduceMax(float v) {
    __shared__ float sm[8];
    #pragma unroll
    for (int o = 16; o > 0; o >>= 1) v = fmaxf(v, __shfl_xor(v, o, 32));
    int w = threadIdx.x >> 5;
    __syncthreads();
    if ((threadIdx.x & 31) == 0) sm[w] = v;
    __syncthreads();
    float r = sm[0];
    #pragma unroll
    for (int i = 1; i < 8; i++) r = fmaxf(r, sm[i]);
    return r;
}

#if HAVE_TDM
typedef __attribute__((ext_vector_type(4))) unsigned int u32x4;
typedef __attribute__((ext_vector_type(8))) int          i32x8;
typedef __attribute__((ext_vector_type(4))) int          i32x4;

// Issue one 2D TDM tile load: rows x 32 bf16 elements, row stride = strideElems,
// destination LDS rows padded to 40 u16 (pad 4 DWORDs after every 16 DWORDs).
__device__ __forceinline__ void tdm_load_tile(unsigned ldsByteAddr,
                                              const unsigned short* g,
                                              unsigned strideElems, unsigned rows)
{
    unsigned long long ga = (unsigned long long)(uintptr_t)g;
    u32x4 g0;
    g0[0] = 1u;                                               // count=1, user mode
    g0[1] = ldsByteAddr;                                      // lds_addr
    g0[2] = (unsigned)ga;                                     // global_addr lo
    g0[3] = ((unsigned)(ga >> 32) & 0x1FFFFFFu) | (2u << 30); // global_addr hi | type=2
    i32x8 g1;
    g1[0] = (1 << 16) | (1 << 20) | (3 << 22) | (3 << 25);    // data_size=2B, pad_en,
                                                              // interval=16dw, amount=4dw
    g1[1] = (int)(32u << 16);                                 // tensor_dim0 = 32
    g1[2] = (int)((rows & 0xFFFFu) << 16);                    // tensor_dim1 lo16
    g1[3] = (int)((rows >> 16) | (32u << 16));                // tensor_dim1 hi | tile_dim0=32
    g1[4] = (int)(rows & 0xFFFFu);                            // tile_dim1=rows, tile_dim2=0
    g1[5] = (int)strideElems;                                 // tensor_dim0_stride lo32
    g1[6] = 0;                                                // stride hi | dim1_stride lo
    g1[7] = 0;
    i32x4 z4 = {0, 0, 0, 0};
    i32x8 z8 = {0, 0, 0, 0, 0, 0, 0, 0};
    __builtin_amdgcn_tensor_load_to_lds(g0, g1, z4, z4, z8, 0);
}
#endif

// 4 WMMAs for one 32-wide K step: wave computes a 32x32 C sub-tile.
__device__ __forceinline__ void mma_step(const unsigned short (*As)[40],
                                         const unsigned short (*Bs)[40],
                                         int wr, int wc, int hi, int l15,
                                         v8f& a00, v8f& a01, v8f& a10, v8f& a11)
{
    FragU fa0, fa1, fb0, fb1;
    const int m0 = wr * 32 + l15, m1 = m0 + 16;
    fa0.u[0] = *(const uint4*)&As[m0][hi * 8];
    fa0.u[1] = *(const uint4*)&As[m0][hi * 8 + 16];
    fa1.u[0] = *(const uint4*)&As[m1][hi * 8];
    fa1.u[1] = *(const uint4*)&As[m1][hi * 8 + 16];
    const int n0 = wc * 32 + l15;
    fb0.u[0] = *(const uint4*)&Bs[n0][hi * 8];
    fb0.u[1] = *(const uint4*)&Bs[n0][hi * 8 + 16];
    fb1.u[0] = *(const uint4*)&Bs[n0 + 16][hi * 8];
    fb1.u[1] = *(const uint4*)&Bs[n0 + 16][hi * 8 + 16];
    a00 = __builtin_amdgcn_wmma_f32_16x16x32_bf16(false, fa0.v, false, fb0.v, (short)0, a00, false, false);
    a01 = __builtin_amdgcn_wmma_f32_16x16x32_bf16(false, fa0.v, false, fb1.v, (short)0, a01, false, false);
    a10 = __builtin_amdgcn_wmma_f32_16x16x32_bf16(false, fa1.v, false, fb0.v, (short)0, a10, false, false);
    a11 = __builtin_amdgcn_wmma_f32_16x16x32_bf16(false, fa1.v, false, fb1.v, (short)0, a11, false, false);
}

// ---------------- bf16 WMMA GEMM --------------------------------------
// C = scale * (A x op(B)) + bias [+relu].  A,B are bf16 (raw u16), C fp32 or bf16.
// BT=true : B is [N,K] row-major (A*B^T);  BT=false: B is [K,N] row-major (A*B).
// Block tile 128x64, 8 waves, 32x32 per wave. M%128==0, N%64==0, K%32==0.
// tdmFlag!=0 (BT only): double-buffered Tensor-Data-Mover loads, wave0 issues.
template<bool BT>
__global__ __launch_bounds__(256) void gemm_bf16(
    const unsigned short* __restrict__ A, int lda, long long sA,
    const unsigned short* __restrict__ B, int ldb, long long sB,
    float* __restrict__ Cf, unsigned short* __restrict__ Ch, int ldc, long long sC,
    int K, const float* __restrict__ bias, float scale, int relu, int tdmFlag)
{
    __shared__ __align__(16) unsigned short As[2][128][40];
    __shared__ __align__(16) unsigned short Bs[2][64][40];

    const int tid  = threadIdx.x;
    const int lane = tid & 31;
    const int wave = tid >> 5;
    const int hi   = lane >> 4;
    const int l15  = lane & 15;
    const int wr   = wave & 3;   // 4 wave-rows of 32
    const int wc   = wave >> 2;  // 2 wave-cols of 32

    const unsigned short* Ab = A + (long long)blockIdx.z * sA;
    const unsigned short* Bb = B + (long long)blockIdx.z * sB;
    const int row0 = blockIdx.x * 128;
    const int col0 = blockIdx.y * 64;

    v8f a00 = {0.f,0.f,0.f,0.f,0.f,0.f,0.f,0.f};
    v8f a01 = a00, a10 = a00, a11 = a00;

#if HAVE_TDM
    if (BT && tdmFlag) {
        const unsigned ldsA0 = (unsigned)(uintptr_t)&As[0][0][0];
        const unsigned ldsA1 = (unsigned)(uintptr_t)&As[1][0][0];
        const unsigned ldsB0 = (unsigned)(uintptr_t)&Bs[0][0][0];
        const unsigned ldsB1 = (unsigned)(uintptr_t)&Bs[1][0][0];
        const unsigned short* Arow = Ab + (long long)row0 * lda;
        const unsigned short* Brow = Bb + (long long)col0 * ldb;
        if (wave == 0) {
            tdm_load_tile(ldsA0, Arow, (unsigned)lda, 128u);
            tdm_load_tile(ldsB0, Brow, (unsigned)ldb, 64u);
        }
        for (int kk = 0; kk < K; kk += 32) {
            const int p = (kk >> 5) & 1;
            if (wave == 0) __builtin_amdgcn_s_wait_tensorcnt(0);
            __syncthreads();                       // buf p ready; buf 1-p free
            if (wave == 0 && kk + 32 < K) {
                tdm_load_tile(p ? ldsA0 : ldsA1, Arow + kk + 32, (unsigned)lda, 128u);
                tdm_load_tile(p ? ldsB0 : ldsB1, Brow + kk + 32, (unsigned)ldb, 64u);
            }
            mma_step(As[p], Bs[p], wr, wc, hi, l15, a00, a01, a10, a11);
        }
    } else
#endif
    {
        (void)tdmFlag;
        for (int kk = 0; kk < K; kk += 32) {
            __syncthreads();
            // A tile 128x32 bf16: 512 16B chunks, 2 per thread
            #pragma unroll
            for (int j = 0; j < 2; j++) {
                int idx = tid + j * 256;
                int r   = idx >> 2;
                int c8  = (idx & 3) * 8;
                *(uint4*)&As[0][r][c8] =
                    *(const uint4*)&Ab[(long long)(row0 + r) * lda + kk + c8];
            }
            if (BT) {      // B[N,K]: 64x32 tile, 256 16B chunks, 1 per thread
                int n  = tid >> 2;
                int c8 = (tid & 3) * 8;
                *(uint4*)&Bs[0][n][c8] =
                    *(const uint4*)&Bb[(long long)(col0 + n) * ldb + kk + c8];
            } else {       // B[K,N]: transpose 32x64 tile into Bs[n][k]
                int k  = tid >> 3;
                int n8 = (tid & 7) * 8;
                Half8 h;
                h.u = *(const uint4*)&Bb[(long long)(kk + k) * ldb + col0 + n8];
                #pragma unroll
                for (int i = 0; i < 8; i++) Bs[0][n8 + i][k] = h.s[i];
            }
            __syncthreads();
            mma_step(As[0], Bs[0], wr, wc, hi, l15, a00, a01, a10, a11);
        }
    }

    // ---- epilogue: scale, +bias, optional relu, fp32 or bf16 store ----
    const int crow = row0 + wr * 32 + hi * 8;
    const int ccol = col0 + wc * 32 + l15;
    const float b0 = bias ? bias[ccol]      : 0.f;
    const float b1 = bias ? bias[ccol + 16] : 0.f;
    float* Cfb = Cf ? Cf + (long long)blockIdx.z * sC : nullptr;
    unsigned short* Chb = Ch ? Ch + (long long)blockIdx.z * sC : nullptr;
    v8f accs[2][2] = {{a00, a01}, {a10, a11}};
    #pragma unroll
    for (int t = 0; t < 2; t++) {
        #pragma unroll
        for (int r = 0; r < 8; r++) {
            const long long rowi = (long long)(crow + t * 16 + r) * ldc;
            float v0 = accs[t][0][r] * scale + b0;
            float v1 = accs[t][1][r] * scale + b1;
            if (relu) { v0 = fmaxf(v0, 0.f); v1 = fmaxf(v1, 0.f); }
            if (Chb) {
                Chb[rowi + ccol]      = f2bf(v0);
                Chb[rowi + ccol + 16] = f2bf(v1);
            } else {
                Cfb[rowi + ccol]      = v0;
                Cfb[rowi + ccol + 16] = v1;
            }
        }
    }
}

// ---------------- fp32 -> bf16 bulk convert (n % 1024 == 0) -----------
__global__ __launch_bounds__(256) void to_bf16(const float* __restrict__ in,
                                               unsigned short* __restrict__ out,
                                               long long n)
{
    long long i = ((long long)blockIdx.x * 256 + threadIdx.x) * 4;
    if (i < n) {
        float4 f = *(const float4*)&in[i];
        uint2 o;
        o.x = (unsigned)f2bf(f.x) | ((unsigned)f2bf(f.y) << 16);
        o.y = (unsigned)f2bf(f.z) | ((unsigned)f2bf(f.w) << 16);
        *(uint2*)&out[i] = o;
    }
}

// ---------------- attention softmax chain -----------------------------
// S: per-batch logits [NHEADS, SEQ, SEQ]. One block per query row q:
// per-head softmax averaged over heads -> causal mask -> +pos -> softmax -> bf16
__global__ __launch_bounds__(256) void attn_softmax(const float* __restrict__ S,
                                                    unsigned short* __restrict__ Pbf)
{
    const int q = blockIdx.x;
    const int tid = threadIdx.x;
    float acc[4] = {0.f, 0.f, 0.f, 0.f};
    for (int h = 0; h < NHEADS; h++) {
        const float* row = S + ((long long)h * SEQ + q) * SEQ;
        float v[4], m = -INFINITY;
        #pragma unroll
        for (int i = 0; i < 4; i++) { v[i] = row[tid + i * 256]; m = fmaxf(m, v[i]); }
        m = blockReduceMax(m);
        float s = 0.f;
        #pragma unroll
        for (int i = 0; i < 4; i++) { v[i] = __expf(v[i] - m); s += v[i]; }
        s = blockReduceSum(s);
        const float inv = 1.f / (s * (float)NHEADS);
        #pragma unroll
        for (int i = 0; i < 4; i++) acc[i] += v[i] * inv;
    }
    float v2[4], m2 = -INFINITY;
    #pragma unroll
    for (int i = 0; i < 4; i++) {
        const int col = tid + i * 256;
        float a = acc[i];
        if (col > q && a != 0.f) a = -INFINITY;
        a += __expf(-fabsf((float)(col - q)) * POS_SCALE);   // POS_WEIGHT = 1
        v2[i] = a; m2 = fmaxf(m2, a);
    }
    m2 = blockReduceMax(m2);
    float s2 = 0.f;
    #pragma unroll
    for (int i = 0; i < 4; i++) { v2[i] = __expf(v2[i] - m2); s2 += v2[i]; }
    s2 = blockReduceSum(s2);
    const float inv2 = 1.f / s2;
    #pragma unroll
    for (int i = 0; i < 4; i++)
        Pbf[(long long)q * SEQ + tid + i * 256] = f2bf(v2[i] * inv2);
}

// ---------------- fused residual + LayerNorm (+ optional bf16 copy) ----
__global__ __launch_bounds__(256) void residual_ln(const float* __restrict__ X,
                                                   const float* __restrict__ R,
                                                   const float* __restrict__ g,
                                                   const float* __restrict__ bt,
                                                   float* __restrict__ Out,
                                                   unsigned short* __restrict__ OutBf)
{
    const long long row = blockIdx.x;
    const int tid = threadIdx.x;
    float v[2];
    #pragma unroll
    for (int i = 0; i < 2; i++) {
        const int e = tid + i * 256;
        float x = X[row * EMB + e];
        if (R) x += R[row * EMB + e];
        v[i] = x;
    }
    const float mean = blockReduceSum(v[0] + v[1]) * (1.f / EMB);
    const float d0 = v[0] - mean, d1 = v[1] - mean;
    const float var = blockReduceSum(d0 * d0 + d1 * d1) * (1.f / EMB);
    const float inv = rsqrtf(var + LNEPS);
    #pragma unroll
    for (int i = 0; i < 2; i++) {
        const int e = tid + i * 256;
        const float o = (v[i] - mean) * inv * g[e] + bt[e];
        Out[row * EMB + e] = o;
        if (OutBf) OutBf[row * EMB + e] = f2bf(o);
    }
}

// ---------------- host orchestration ----------------------------------
extern "C" void kernel_launch(void* const* d_in, const int* in_sizes, int n_in,
                              void* d_out, int out_size, void* d_ws, size_t ws_size,
                              hipStream_t stream)
{
    (void)in_sizes; (void)n_in; (void)out_size; (void)ws_size;

    const float* in   = (const float*)d_in[0];
    const float* Wqkv = (const float*)d_in[1];
    const float* bqkv = (const float*)d_in[2];
    const float* W1   = (const float*)d_in[3];
    const float* b1   = (const float*)d_in[4];
    const float* W2   = (const float*)d_in[5];
    const float* b2   = (const float*)d_in[6];
    const float* g1   = (const float*)d_in[7];
    const float* bt1  = (const float*)d_in[8];
    const float* g2   = (const float*)d_in[9];
    const float* bt2  = (const float*)d_in[10];
    const float* g3   = (const float*)d_in[11];
    const float* bt3  = (const float*)d_in[12];

    const long long nX   = (long long)NB * SEQ * EMB;       //  4M
    const long long nS   = (long long)NHEADS * SEQ * SEQ;   //  8M
    const long long nQK  = (long long)NB * SEQ * 2 * EMB;   //  8M (q,k only — v unused)
    const long long nH   = (long long)NB * SEQ * FFDIM;     // 16M
    const long long nWq  = (long long)2 * EMB * EMB;        // 512K
    const long long nW1  = (long long)FFDIM * EMB;          //  1M

    float* X = (float*)d_ws;
    float* Y = X + nX;
    float* S = Y + nX;
    unsigned short* Xbf   = (unsigned short*)(S + nS);
    unsigned short* QKbf  = Xbf + nX;
    unsigned short* Pbf   = QKbf + nQK;                 // softmax probs (1M)
    unsigned short* Hbf   = Pbf + (long long)SEQ * SEQ;
    unsigned short* cWq   = Hbf + nH;
    unsigned short* cW1   = cWq + nWq;
    unsigned short* cW2   = cW1 + nW1;

    (void)hipMemcpyAsync(X, in, nX * sizeof(float), hipMemcpyDeviceToDevice, stream);
    to_bf16<<<nX / 1024, 256, 0, stream>>>(X, Xbf, nX);

    const int M = NB * SEQ;                   // 8192 token rows
    const float inv_sqrt_hd = 0.125f;         // 1/sqrt(64)

    for (int d = 0; d < DEPTH_; d++) {
        const float* Wq_d = Wqkv + (long long)d * 3 * EMB * EMB;
        const float* bq_d = bqkv + (long long)d * 3 * EMB;
        const float* W1_d = W1 + (long long)d * FFDIM * EMB;
        const float* b1_d = b1 + (long long)d * FFDIM;
        const float* W2_d = W2 + (long long)d * EMB * FFDIM;
        const float* b2_d = b2 + (long long)d * EMB;

        // convert this layer's weights to bf16 (q,k rows of Wqkv only)
        to_bf16<<<nWq / 1024, 256, 0, stream>>>(Wq_d, cWq, nWq);
        to_bf16<<<nW1 / 1024, 256, 0, stream>>>(W1_d, cW1, nW1);
        to_bf16<<<nW1 / 1024, 256, 0, stream>>>(W2_d, cW2, nW1);

        // [q|k] = x @ Wqkv[0:2E]^T + bqkv[0:2E]  -> bf16 [8192, 1024]
        gemm_bf16<true><<<dim3(M / 128, (2 * EMB) / 64, 1), 256, 0, stream>>>(
            Xbf, EMB, 0, cWq, EMB, 0, nullptr, QKbf, 2 * EMB, 0,
            EMB, bq_d, 1.f, 0, 1);

        for (int b = 0; b < NB; b++) {
            const unsigned short* qb = QKbf + (long long)b * SEQ * 2 * EMB;
            const unsigned short* kb = qb + EMB;
            // logits[h] = (q_h @ k_h^T)/sqrt(hd), batched over heads via z
            gemm_bf16<true><<<dim3(SEQ / 128, SEQ / 64, NHEADS), 256, 0, stream>>>(
                qb, 2 * EMB, HD, kb, 2 * EMB, HD, S, nullptr, SEQ,
                (long long)SEQ * SEQ, HD, nullptr, inv_sqrt_hd, 0, 1);
            // head-avg softmax -> mask -> +pos -> softmax -> bf16 probs
            attn_softmax<<<SEQ, 256, 0, stream>>>(S, Pbf);
            // sa = P @ x   (value is the raw layer-input x, still in Xbf)
            gemm_bf16<false><<<dim3(SEQ / 128, EMB / 64, 1), 256, 0, stream>>>(
                Pbf, SEQ, 0, Xbf + (long long)b * SEQ * EMB, EMB, 0,
                Y + (long long)b * SEQ * EMB, nullptr, EMB, 0,
                SEQ, nullptr, 1.f, 0, 0);
        }

        residual_ln<<<M, 256, 0, stream>>>(X, Y, g1 + d * EMB, bt1 + d * EMB, X, nullptr);
        residual_ln<<<M, 256, 0, stream>>>(X, nullptr, g2 + d * EMB, bt2 + d * EMB, X, Xbf);

        // FFN: h = relu(x @ W1^T + b1) -> bf16; ff = h @ W2^T + b2 -> fp32
        gemm_bf16<true><<<dim3(M / 128, FFDIM / 64, 1), 256, 0, stream>>>(
            Xbf, EMB, 0, cW1, EMB, 0, nullptr, Hbf, FFDIM, 0,
            EMB, b1_d, 1.f, 1, 1);
        gemm_bf16<true><<<dim3(M / 128, EMB / 64, 1), 256, 0, stream>>>(
            Hbf, FFDIM, 0, cW2, FFDIM, 0, Y, nullptr, EMB, 0,
            FFDIM, b2_d, 1.f, 0, 1);

        residual_ln<<<M, 256, 0, stream>>>(X, Y, g3 + d * EMB, bt3 + d * EMB, X, Xbf);
    }

    (void)hipMemcpyAsync(d_out, X, nX * sizeof(float), hipMemcpyDeviceToDevice, stream);
}
